// ST_GAGCN_39737037423328
// MI455X (gfx1250) — compile-verified, matrix operand order checked by reading, not run on hardware
//
#include <hip/hip_runtime.h>
#include <hip/hip_bf16.h>

typedef __attribute__((ext_vector_type(16))) _Float16 v16h;
typedef __attribute__((ext_vector_type(8)))  _Float16 v8h;
typedef __attribute__((ext_vector_type(8)))  float    v8f;
typedef __attribute__((ext_vector_type(4)))  float    v4f;
typedef __attribute__((ext_vector_type(16))) float    v16f;

#define NEGC (-9.0e15f)

__device__ __forceinline__ v8f wmma16(v16h a, v16h b, v8f c) {
    return __builtin_amdgcn_wmma_f32_16x16x32_f16(false, a, false, b, (short)0, c,
                                                  false, false);
}

// Fragment load: lane's 16 halfs are two contiguous 8-half runs at
// p[half*8 .. +8) and p[half*8+16 .. +8)  ->  2 x 16B vector loads.
__device__ __forceinline__ v16h ldfrag(const _Float16* p, int half) {
    v8h lo = *(const v8h*)(p + half * 8);
    v8h hi = *(const v8h*)(p + half * 8 + 16);
    return __builtin_shufflevector(lo, hi, 0, 1, 2, 3, 4, 5, 6, 7, 8, 9, 10, 11, 12,
                                   13, 14, 15);
}

// 16 f32 values along k (two contiguous runs) -> register vector (no stack).
__device__ __forceinline__ v16f ld16fv(const float* src, int half) {
    v4f a = *(const v4f*)(src + half * 8);
    v4f b = *(const v4f*)(src + half * 8 + 4);
    v4f c = *(const v4f*)(src + half * 8 + 16);
    v4f d = *(const v4f*)(src + half * 8 + 20);
    v8f lo = __builtin_shufflevector(a, b, 0, 1, 2, 3, 4, 5, 6, 7);
    v8f hi = __builtin_shufflevector(c, d, 0, 1, 2, 3, 4, 5, 6, 7);
    return __builtin_shufflevector(lo, hi, 0, 1, 2, 3, 4, 5, 6, 7, 8, 9, 10, 11, 12,
                                   13, 14, 15);
}

// ---------------------------------------------------------------------------
// Weight conversion to f16, stored N-MAJOR (w16t[n*K + k]).
// segments: [0)=sp 64x256, [16384)=qkv 384x128(copy), [65536)=oW 128x128(copy),
//           [81920)=o1 64x128, [90112)=o2 120x64 ; total 97792
// ---------------------------------------------------------------------------
__global__ void k_cvt(const float* __restrict__ sp_w, const float* __restrict__ in_w,
                      const float* __restrict__ out_w, const float* __restrict__ o1_w,
                      const float* __restrict__ o2_w, _Float16* __restrict__ w16) {
    int t = blockIdx.x * 256 + threadIdx.x;
    if (t >= 97792) return;
    if (t < 16384) {
        int nn = t / 256, kk = t % 256;
        w16[t] = (_Float16)sp_w[kk * 64 + nn];
    } else if (t < 65536) {
        w16[t] = (_Float16)in_w[t - 16384];
    } else if (t < 81920) {
        w16[t] = (_Float16)out_w[t - 65536];
    } else if (t < 90112) {
        int e = t - 81920; int nn = e / 128, kk = e % 128;
        w16[t] = (_Float16)o1_w[kk * 64 + nn];
    } else {
        int e = t - 90112; int nn = e / 64, kk = e % 64;
        w16[t] = (_Float16)o2_w[kk * 120 + nn];
    }
}

// ---------------------------------------------------------------------------
// Wh = x_last @ gat_W per (b,k,n); stores Wh TRANSPOSED: wht[bk][h][j=n].
// ---------------------------------------------------------------------------
__global__ void k_wh(const float* __restrict__ x, const float* __restrict__ gat_W,
                     const float* __restrict__ a1, const float* __restrict__ a2,
                     _Float16* __restrict__ wht, float* __restrict__ f1,
                     float* __restrict__ f2) {
    int id = blockIdx.x;
    int n = id & 1023, k = (id >> 10) & 3, b = id >> 12;
    __shared__ float xs[16];
    __shared__ float r1[64], r2[64];
    int h = threadIdx.x;
    if (h < 16) xs[h] = x[(((size_t)(b * 12 + 11) * 1024 + n) * 16) + h];
    __syncthreads();
    const float* W = gat_W + (size_t)(k * 16) * 64 + h;
    float acc = 0.f;
#pragma unroll
    for (int f = 0; f < 16; ++f) acc += xs[f] * W[(size_t)f * 64];
    wht[((size_t)(b * 4 + k) * 64 + h) * 1024 + n] = (_Float16)acc;
    r1[h] = acc * a1[k * 64 + h];
    r2[h] = acc * a2[k * 64 + h];
    __syncthreads();
    for (int s = 32; s > 0; s >>= 1) {
        if (h < s) { r1[h] += r1[h + s]; r2[h] += r2[h + s]; }
        __syncthreads();
    }
    if (h == 0) {
        f1[(b * 4 + k) * 1024 + n] = r1[0];
        f2[(b * 4 + k) * 1024 + n] = r2[0];
    }
}

// ---------------------------------------------------------------------------
// GAT masked-softmax row statistics (branchless online). One wave per row.
// ---------------------------------------------------------------------------
__global__ void k_gat_stats(const float* __restrict__ adj, const float* __restrict__ f1,
                            const float* __restrict__ f2, float* __restrict__ rmax,
                            float* __restrict__ rsum) {
    int wid = threadIdx.x >> 5, lane = threadIdx.x & 31;
    int row = blockIdx.x * 4 + wid;
    int i = row & 1023, k = (row >> 10) & 3, b = row >> 12;
    int bk = b * 4 + k;
    float f1v = f1[bk * 1024 + i];
    const float* arow = adj + ((size_t)b * 1024 + i) * 1024;
    const float* f2r = f2 + bk * 1024;
    float m = -3.0e38f, s = 0.f;
    for (int j = lane; j < 1024; j += 32) {
        float t = f1v + f2r[j];
        float lr = fmaxf(t, 0.2f * t);                  // leaky_relu, branchless
        float e = (arow[j] > 0.f) ? lr : NEGC;          // cndmask
        float mn = fmaxf(m, e);
        s = s * __expf(m - mn) + __expf(e - mn);
        m = mn;
    }
    for (int d = 1; d < 32; d <<= 1) {
        float mo = __shfl_xor(m, d, 32);
        float so = __shfl_xor(s, d, 32);
        float mn = fmaxf(m, mo);
        s = s * __expf(m - mn) + so * __expf(mo - mn);
        m = mn;
    }
    if (lane == 0) { rmax[row] = m; rsum[row] = s; }
}

// ---------------------------------------------------------------------------
// GAT aggregation via WMMA. Branchless A-frag from register-vector adj/f2
// runs; B-frags are 2x16B global loads from transposed Wh.
// ---------------------------------------------------------------------------
__global__ void k_gat_agg(const float* __restrict__ adj, const float* __restrict__ f1,
                          const float* __restrict__ f2, const float* __restrict__ rmax,
                          const float* __restrict__ rsum,
                          const _Float16* __restrict__ wht,
                          _Float16* __restrict__ cat16) {
    int id = blockIdx.x;
    int it = id & 63, k = (id >> 6) & 3, b = id >> 8;
    int bk = b * 4 + k;
    int i0 = it * 16;
    int lane = threadIdx.x, half = lane >> 4, lm = lane & 15;
    float f1v = f1[bk * 1024 + i0 + lm];
    float mrow = rmax[bk * 1024 + i0 + lm];
    const float* arow = adj + ((size_t)b * 1024 + (i0 + lm)) * 1024;
    const float* f2r = f2 + bk * 1024;
    const _Float16* whb = wht + (size_t)bk * 65536;   // [h][j], 64x1024
    v8f acc[4] = {};
    for (int j0 = 0; j0 < 1024; j0 += 32) {
        v16f adjv = ld16fv(arow + j0, half);
        v16f f2v  = ld16fv(f2r + j0, half);
        v16h a;
#pragma unroll
        for (int i = 0; i < 16; ++i) {
            float t = f1v + f2v[i];
            float lr = fmaxf(t, 0.2f * t);
            float e = (adjv[i] > 0.f) ? lr : NEGC;
            a[i] = (_Float16)__expf(e - mrow);
        }
#pragma unroll
        for (int t = 0; t < 4; ++t) {
            v16h bf = ldfrag(whb + (size_t)(t * 16 + lm) * 1024 + j0, half);
            acc[t] = wmma16(a, bf, acc[t]);
        }
    }
    float linv[8];
#pragma unroll
    for (int v = 0; v < 8; ++v)
        linv[v] = 1.f / rsum[bk * 1024 + i0 + v + half * 8];
#pragma unroll
    for (int t = 0; t < 4; ++t)
#pragma unroll
        for (int v = 0; v < 8; ++v) {
            int ig = i0 + v + half * 8;
            float val = acc[t][v] * linv[v];
            val = val > 0.f ? val : (__expf(val) - 1.f);   // ELU
            cat16[((size_t)(b * 1024 + ig)) * 256 + k * 64 + t * 16 + lm] = (_Float16)val;
        }
}

// ---------------------------------------------------------------------------
// Gated TCN (taps t={7,9,11}). grid = B*N, block = 64.
// ---------------------------------------------------------------------------
__global__ void k_tcn(const float* __restrict__ x, const float* __restrict__ w1,
                      const float* __restrict__ b1, const float* __restrict__ w2,
                      const float* __restrict__ b2, float* __restrict__ fi,
                      _Float16* __restrict__ fi16) {
    int bn = blockIdx.x;
    int n = bn & 1023, b = bn >> 10;
    __shared__ float xs[48];
    int c = threadIdx.x;
    if (c < 48) {
        int tt = c / 16, f = c % 16;
        int t = 7 + tt * 2;
        xs[c] = x[((size_t)(b * 12 + t) * 1024 + n) * 16 + f];
    }
    __syncthreads();
    float p = b1[c], q = b2[c];
#pragma unroll
    for (int kk = 0; kk < 3; ++kk)
#pragma unroll
        for (int f = 0; f < 16; ++f) {
            float xv = xs[kk * 16 + f];
            p += xv * w1[(c * 16 + f) * 3 + kk];
            q += xv * w2[(c * 16 + f) * 3 + kk];
        }
    float val = tanhf(p) * (1.f / (1.f + __expf(-q)));
    size_t idx = (size_t)bn * 128 + 64 + c;
    fi[idx] = val;
    fi16[idx] = (_Float16)val;
}

// ---------------------------------------------------------------------------
// MHA attention. Pass 1: WMMA scores, max-only reduction (1 fmax/elem, lane
// merge once at end). Pass 2: WMMA scores -> p=exp(s-M) (1 exp/elem), denom
// accumulated per-lane for free, P/V transposed through small LDS, 2 WMMAs.
// ---------------------------------------------------------------------------
__global__ void k_mha(const _Float16* __restrict__ qkv16, _Float16* __restrict__ attn16) {
    int id = blockIdx.x;
    int it = id & 63, h = (id >> 6) & 3, b = id >> 8;
    int i0 = it * 16;
    int lane = threadIdx.x, half = lane >> 4, lm = lane & 15;
    const float scale = 0.17677669529663687f;   // 1/sqrt(32)
    __shared__ _Float16 pbuf[16 * 32];          // [i][j]
    __shared__ _Float16 vbuf[32 * 32];          // [d][j]

    v16h aq = ldfrag(qkv16 + ((size_t)(b * 1024) + i0 + lm) * 384 + h * 32, half);

    // pass 1: row max only
    float M[8];
#pragma unroll
    for (int v = 0; v < 8; ++v) M[v] = -3.0e38f;
    for (int j0 = 0; j0 < 1024; j0 += 16) {
        v16h bk =
            ldfrag(qkv16 + ((size_t)(b * 1024) + j0 + lm) * 384 + 128 + h * 32, half);
        v8f s = {};
        s = wmma16(aq, bk, s);
#pragma unroll
        for (int v = 0; v < 8; ++v) M[v] = fmaxf(M[v], s[v] * scale);
    }
#pragma unroll
    for (int v = 0; v < 8; ++v)
        for (int d = 1; d < 16; d <<= 1) M[v] = fmaxf(M[v], __shfl_xor(M[v], d, 32));

    // pass 2: p = exp(s - M); denom accumulated per-lane; out = P @ V
    float S[8];
#pragma unroll
    for (int v = 0; v < 8; ++v) S[v] = 0.f;
    v8f acc0 = {}, acc1 = {};
    for (int j0 = 0; j0 < 1024; j0 += 32) {
#pragma unroll
        for (int tt = 0; tt < 2; ++tt) {
            v16h bk = ldfrag(
                qkv16 + ((size_t)(b * 1024) + j0 + tt * 16 + lm) * 384 + 128 + h * 32,
                half);
            v8f s = {};
            s = wmma16(aq, bk, s);
#pragma unroll
            for (int v = 0; v < 8; ++v) {
                float p = __expf(s[v] * scale - M[v]);
                S[v] += p;
                pbuf[(v + half * 8) * 32 + tt * 16 + lm] = (_Float16)p;
            }
        }
        {   // stage V rows j0..j0+31 transposed into vbuf[d][j]
            _Float16 vrow[32];
            const int4* src =
                (const int4*)(qkv16 + ((size_t)(b * 1024) + j0 + lane) * 384 + 256 +
                              h * 32);
#pragma unroll
            for (int r = 0; r < 4; ++r) *(int4*)&vrow[r * 8] = src[r];
#pragma unroll
            for (int d = 0; d < 32; ++d) vbuf[d * 32 + lane] = vrow[d];
        }
        __syncthreads();
        v16h ap  = ldfrag(pbuf + lm * 32, half);
        v16h bv0 = ldfrag(vbuf + lm * 32, half);
        v16h bv1 = ldfrag(vbuf + (16 + lm) * 32, half);
        acc0 = wmma16(ap, bv0, acc0);
        acc1 = wmma16(ap, bv1, acc1);
        __syncthreads();
    }
#pragma unroll
    for (int v = 0; v < 8; ++v)
        for (int d = 1; d < 16; d <<= 1) S[v] += __shfl_xor(S[v], d, 32);
#pragma unroll
    for (int v = 0; v < 8; ++v) {
        int ig = i0 + v + half * 8;
        float inv = 1.f / S[v];
        attn16[((size_t)(b * 1024) + ig) * 128 + h * 32 + lm] = (_Float16)(acc0[v] * inv);
        attn16[((size_t)(b * 1024) + ig) * 128 + h * 32 + 16 + lm] =
            (_Float16)(acc1[v] * inv);
    }
}

// ---------------------------------------------------------------------------
// Generic WMMA GEMM, no LDS: A row-major MxK, B stored N-major [n][k].
// ---------------------------------------------------------------------------
#define EP_QKV     0
#define EP_SPATIAL 1
#define EP_OPROJ   2
#define EP_RELU    3
#define EP_OUT     4

__global__ void k_gemm(const _Float16* __restrict__ A, const _Float16* __restrict__ Bt,
                       const float* __restrict__ bias, int Ktot, int Ntot, int mode,
                       const float* __restrict__ auxF32, _Float16* __restrict__ dst16,
                       float* __restrict__ dst32) {
    int tid = threadIdx.x;
    int wid = tid >> 5, lane = tid & 31, half = lane >> 4, lm = lane & 15;
    int m0 = blockIdx.x * 64;
    int n0 = blockIdx.y * 64;
    const _Float16* arow = A + (size_t)(m0 + wid * 16 + lm) * Ktot;
    v8f acc[4] = {};
    for (int k0 = 0; k0 < Ktot; k0 += 32) {
        v16h a = ldfrag(arow + k0, half);
#pragma unroll
        for (int t = 0; t < 4; ++t) {
            int gn = n0 + t * 16 + lm;
            int gnl = gn < Ntot ? gn : (Ntot - 1);
            v16h bf = ldfrag(Bt + (size_t)gnl * Ktot + k0, half);
            acc[t] = wmma16(a, bf, acc[t]);
        }
    }
#pragma unroll
    for (int t = 0; t < 4; ++t) {
        int gn = n0 + t * 16 + lm;
        if (gn >= Ntot) continue;
        float bv = bias ? bias[gn] : 0.f;
#pragma unroll
        for (int v = 0; v < 8; ++v) {
            int gm = m0 + wid * 16 + v + half * 8;
            float c = acc[t][v] + bv;
            if (mode == EP_QKV) {
                dst16[(size_t)gm * Ntot + gn] = (_Float16)c;
            } else if (mode == EP_SPATIAL) {
                dst32[(size_t)gm * 128 + gn] = c;
                dst16[(size_t)gm * 128 + gn] = (_Float16)c;
            } else if (mode == EP_OPROJ) {
                c += auxF32[(size_t)gm * 128 + gn];
                dst16[(size_t)gm * 128 + gn] = (_Float16)c;
            } else if (mode == EP_RELU) {
                dst16[(size_t)gm * 64 + gn] = (_Float16)(c > 0.f ? c : 0.f);
            } else {   // EP_OUT : (B,N,P,2) -> (B,P,N,2)
                int bb = gm >> 10, nn = gm & 1023;
                int p = gn >> 1, ch = gn & 1;
                dst32[(((size_t)(bb * 60) + p) * 1024 + nn) * 2 + ch] = c;
            }
        }
    }
}

// ---------------------------------------------------------------------------
extern "C" void kernel_launch(void* const* d_in, const int* in_sizes, int n_in,
                              void* d_out, int out_size, void* d_ws, size_t ws_size,
                              hipStream_t stream) {
    (void)in_sizes; (void)n_in; (void)out_size; (void)ws_size;
    const float* x      = (const float*)d_in[0];
    const float* adj    = (const float*)d_in[1];
    const float* gat_W  = (const float*)d_in[2];
    const float* gat_a1 = (const float*)d_in[3];
    const float* gat_a2 = (const float*)d_in[4];
    const float* sp_w   = (const float*)d_in[5];
    const float* sp_b   = (const float*)d_in[6];
    const float* c1w    = (const float*)d_in[7];
    const float* c1b    = (const float*)d_in[8];
    const float* c2w    = (const float*)d_in[9];
    const float* c2b    = (const float*)d_in[10];
    const float* in_w   = (const float*)d_in[11];
    const float* in_b   = (const float*)d_in[12];
    const float* ow     = (const float*)d_in[13];
    const float* ob     = (const float*)d_in[14];
    const float* o1w    = (const float*)d_in[15];
    const float* o1b    = (const float*)d_in[16];
    const float* o2w    = (const float*)d_in[17];
    const float* o2b    = (const float*)d_in[18];
    float* out = (float*)d_out;

    char* ws = (char*)d_ws;
    size_t off = 0;
    auto alloc = [&](size_t bytes) -> void* {
        void* p = ws + off;
        off = (off + bytes + 255) & ~(size_t)255;
        return p;
    };
    _Float16* wht    = (_Float16*)alloc(2097152 * 2);     // B*K*H*N (transposed Wh)
    float*    f1     = (float*)alloc(32768 * 4);
    float*    f2     = (float*)alloc(32768 * 4);
    float*    rmax   = (float*)alloc(32768 * 4);
    float*    rsum   = (float*)alloc(32768 * 4);
    _Float16* cat16  = (_Float16*)alloc(2097152 * 2);     // B*N*256
    float*    fi     = (float*)alloc(1048576 * 4);        // B*N*128 f32
    _Float16* fi16   = (_Float16*)alloc(1048576 * 2);
    _Float16* qkv16  = (_Float16*)alloc(3145728 * 2);     // B*N*384
    _Float16* attn16 = (_Float16*)alloc(1048576 * 2);
    _Float16* fo16   = (_Float16*)alloc(1048576 * 2);
    _Float16* hid16  = (_Float16*)alloc(524288 * 2);
    _Float16* w16    = (_Float16*)alloc(97792 * 2);

    k_cvt<<<dim3(382), dim3(256), 0, stream>>>(sp_w, in_w, ow, o1w, o2w, w16);
    k_wh<<<dim3(32768), dim3(64), 0, stream>>>(x, gat_W, gat_a1, gat_a2, wht, f1, f2);
    k_gat_stats<<<dim3(8192), dim3(128), 0, stream>>>(adj, f1, f2, rmax, rsum);
    k_gat_agg<<<dim3(2048), dim3(32), 0, stream>>>(adj, f1, f2, rmax, rsum, wht, cat16);
    k_gemm<<<dim3(128, 1), dim3(128), 0, stream>>>(cat16, w16, sp_b, 256, 64,
                                                   EP_SPATIAL, nullptr, fi16, fi);
    k_tcn<<<dim3(8192), dim3(64), 0, stream>>>(x, c1w, c1b, c2w, c2b, fi, fi16);
    k_gemm<<<dim3(128, 6), dim3(128), 0, stream>>>(fi16, w16 + 16384, in_b, 128, 384,
                                                   EP_QKV, nullptr, qkv16, nullptr);
    k_mha<<<dim3(2048), dim3(32), 0, stream>>>(qkv16, attn16);
    k_gemm<<<dim3(128, 2), dim3(128), 0, stream>>>(attn16, w16 + 65536, ob, 128, 128,
                                                   EP_OPROJ, fi, fo16, nullptr);
    k_gemm<<<dim3(128, 1), dim3(128), 0, stream>>>(fo16, w16 + 81920, o1b, 128, 64,
                                                   EP_RELU, nullptr, hid16, nullptr);
    k_gemm<<<dim3(128, 2), dim3(128), 0, stream>>>(hid16, w16 + 90112, o2b, 64, 120,
                                                   EP_OUT, nullptr, nullptr, out);
}
